// Predictor_50036368998590
// MI455X (gfx1250) — matrix operand order repaired and checked by reading, not compiled, not run on hardware
//
#include <hip/hip_runtime.h>
#include <hip/hip_bf16.h>

// ---------------------------------------------------------------------------
// Hierarchical GRU / ACT model for MI455X (gfx1250), wave32 + WMMA f16.
//
// Shapes: B=64, T=256, D_IN=256, H=512, A=128, DEPTH=4.
// All GEMMs use v_wmma_f32_16x16x32_f16 (f16 in, f32 acc).
// Weights are repacked once per call into f16, N-major ("B^T") layout so a
// B-fragment is one contiguous 32B load per lane (contiguous k order, same as A).
// Per depth, one persistent 16-wave workgroup runs the T=256 recurrence with
// h kept in LDS; GEMM1 = [xt|h] @ [[Wz Wr Wa1];[Uz Ur Ua1]]  (K=1024,N=1152),
// GEMM2 = [xt|r*h] @ [[Wh];[Uh]] (K=1024,N=512), GRU math fused in epilogues.
// NEW: next step's xt tile is staged global->LDS with the CDNA5 async DMA
// (global_load_async_to_lds_b128 / ASYNCcnt), issued right after the last
// reader barrier so the copy overlaps the P4 output stores.
// ---------------------------------------------------------------------------

typedef __attribute__((ext_vector_type(16))) _Float16 v16h;
typedef __attribute__((ext_vector_type(8)))  _Float16 v8h;
typedef __attribute__((ext_vector_type(8)))  float    v8f;

#define WMMA_F16(acc, a, b) \
  (acc) = __builtin_amdgcn_wmma_f32_16x16x32_f16(false, (a), false, (b), (short)0, (acc), false, false)

// --------- problem constants ---------
static constexpr int Bn   = 64;
static constexpr int Tn   = 256;
static constexpr int DIN  = 256;
static constexpr int Hn   = 512;
static constexpr int An   = 128;
static constexpr int NDEP = 4;
static constexpr int K1   = 2 * Hn;          // 1024 stacked K for [x|h]
static constexpr int N1   = 2 * Hn + An;     // 1152 (z | r | pol_h)
static constexpr int N2   = Hn;              // 512  (h_cand)

// --------- d_out layout (float elements), reference return order ---------
static constexpr size_t OUT_OUTPUT = 0;                                   // (B,H)
static constexpr size_t OUT_ACTION = OUT_OUTPUT + (size_t)Bn * Hn;        // (B,D,T)
static constexpr size_t OUT_AC     = OUT_ACTION + (size_t)Bn * NDEP * Tn; // (B,D,T)
static constexpr size_t OUT_PIX    = OUT_AC + (size_t)Bn * NDEP * Tn;     // (B,D,T,H)
static constexpr size_t OUT_PIH    = OUT_PIX + (size_t)Bn * NDEP * Tn * Hn;
static constexpr size_t OUT_POL    = OUT_PIH + (size_t)Bn * NDEP * Tn * Hn; // (B,D,T,2)

// --------- workspace layout (bytes, 256-aligned) ---------
static constexpr size_t WS_WU1T = 0;                               // (N1 x K1) f16
static constexpr size_t WS_WU2T = WS_WU1T + (size_t)N1 * K1 * 2;   // (N2 x K1) f16
static constexpr size_t WS_WET  = WS_WU2T + (size_t)N2 * K1 * 2;   // (H x DIN) f16
static constexpr size_t WS_XH   = WS_WET + (size_t)Hn * DIN * 2;   // x f16 (B,T,DIN)
static constexpr size_t WS_XD0  = WS_XH + (size_t)Bn * Tn * DIN * 2; // (T,B,H) f16
static constexpr size_t WS_XD1  = WS_XD0 + (size_t)Tn * Bn * Hn * 2;
static constexpr size_t WS_ACT0 = WS_XD1 + (size_t)Tn * Bn * Hn * 2; // (T,B) f32
static constexpr size_t WS_ACT1 = WS_ACT0 + (size_t)Tn * Bn * 4;
static constexpr size_t WS_DM0  = WS_ACT1 + (size_t)Tn * Bn * 4;
static constexpr size_t WS_DM1  = WS_DM0 + (size_t)Tn * Bn * 4;

// --------- fragment loaders (contiguous-k layouts, A and B alike) ---------
__device__ __forceinline__ v16h load_frag_a(const _Float16* base, int stride,
                                            int m0, int k0, int lane) {
  // A 16x32 f16: lane&15 = row; 16 contiguous K at k0+(lane>>4)*16.
  const _Float16* p = base + (size_t)(m0 + (lane & 15)) * stride + k0 + ((lane >> 4) << 4);
  union { v16h v; v8h h[2]; } u;
  u.h[0] = *(const v8h*)(p);
  u.h[1] = *(const v8h*)(p + 8);
  return u.v;
}

__device__ __forceinline__ v16h load_frag_b(const _Float16* wt, int stride,
                                            int n0, int k0, int lane) {
  // B 32x16 f16 from N-major (transposed) weights: lane&15 = col, 16 contig K.
  const _Float16* p = wt + (size_t)(n0 + (lane & 15)) * stride + k0 + ((lane >> 4) << 4);
  return *(const v16h*)(p);
}

__device__ __forceinline__ float hsig_f(float v) {
  return fminf(fmaxf(0.2f * v + 0.5f, 0.f), 1.f);
}

// ---------------------------------------------------------------------------
// Prep kernels
// ---------------------------------------------------------------------------
__global__ void cvt_x_kernel(const float* __restrict__ x, _Float16* __restrict__ xh, int n) {
  for (int i = blockIdx.x * blockDim.x + threadIdx.x; i < n; i += gridDim.x * blockDim.x)
    xh[i] = (_Float16)x[i];
}

__global__ void build_w_kernel(const float* __restrict__ W, const float* __restrict__ U,
                               const float* __restrict__ Wa1, const float* __restrict__ Ua1,
                               const float* __restrict__ Wemb,
                               _Float16* __restrict__ WU1t, _Float16* __restrict__ WU2t,
                               _Float16* __restrict__ Wet) {
  const int n1 = N1 * K1;              // 1179648
  const int n2 = n1 + N2 * K1;         // +524288
  const int n3 = n2 + Hn * DIN;        // +131072
  for (int i = blockIdx.x * blockDim.x + threadIdx.x; i < n3; i += gridDim.x * blockDim.x) {
    if (i < n1) {
      int n = i >> 10, k = i & 1023;
      float v;
      if (n < 2 * Hn)  // z,r columns: W/U column index == n
        v = (k < Hn) ? W[(size_t)k * (3 * Hn) + n] : U[(size_t)(k - Hn) * (3 * Hn) + n];
      else {
        int a = n - 2 * Hn;
        v = (k < Hn) ? Wa1[(size_t)k * An + a] : Ua1[(size_t)(k - Hn) * An + a];
      }
      WU1t[i] = (_Float16)v;
    } else if (i < n2) {
      int j = i - n1;
      int n = j >> 10, k = j & 1023;
      int col = 2 * Hn + n;  // Wh/Uh columns
      float v = (k < Hn) ? W[(size_t)k * (3 * Hn) + col] : U[(size_t)(k - Hn) * (3 * Hn) + col];
      WU2t[j] = (_Float16)v;
    } else {
      int j = i - n2;
      int n = j >> 8, k = j & 255;
      Wet[j] = (_Float16)Wemb[(size_t)k * Hn + n];
    }
  }
}

// ---------------------------------------------------------------------------
// Embedding GEMM: xe[t,b,:] = x[b,t,:] @ W_emb + b_emb  (WMMA, fully parallel)
// Writes xe as f16 (T,B,H) for depth 0 and as f32 into d_out pix[:,0,:,:].
// ---------------------------------------------------------------------------
__global__ void embed_kernel(const _Float16* __restrict__ xh, const _Float16* __restrict__ Wet,
                             const float* __restrict__ b_emb,
                             _Float16* __restrict__ xe16, float* __restrict__ dout) {
  const int lane = threadIdx.x & 31;
  const int g = blockIdx.x * (blockDim.x >> 5) + (threadIdx.x >> 5);
  const int mt = g >> 5;          // 1024 row tiles of (b*T+t)
  const int nt = g & 31;          // 32 col tiles of H
  const int r0 = mt * 16, n0 = nt * 16;
  v8f acc = {};
#pragma unroll
  for (int kt = 0; kt < DIN / 32; ++kt) {
    v16h a = load_frag_a(xh, DIN, r0, kt * 32, lane);
    v16h b = load_frag_b(Wet, DIN, n0, kt * 32, lane);
    WMMA_F16(acc, a, b);
  }
  const int col = n0 + (lane & 15);
  const int rbase = r0 + ((lane >> 4) << 3);
  const float bias = b_emb[col];
#pragma unroll
  for (int j = 0; j < 8; ++j) {
    int row = rbase + j;
    int bb = row >> 8, tt = row & 255;   // row = b*T + t
    float v = acc[j] + bias;
    xe16[((size_t)tt * Bn + bb) * Hn + col] = (_Float16)v;
    dout[OUT_PIX + ((size_t)(bb * NDEP) * Tn + tt) * Hn + col] = v;  // pix depth 0
  }
}

// ---------------------------------------------------------------------------
// One depth level: T=256 sequential steps, single 16-wave workgroup.
// Dynamic LDS: A1 [xt|h] f16 (64x1024) | RH f16 (64x512) | ZB f16 (64x512)
//              | POLH f32 (64x128) | per-batch coef arrays.
// ---------------------------------------------------------------------------
static constexpr size_t SEQ_LDS =
    (size_t)Bn * K1 * 2 + (size_t)Bn * Hn * 2 + (size_t)Bn * Hn * 2 +
    (size_t)Bn * An * 4 + 6 * Bn * 4;  // = 296448 bytes (< 320KB WGP LDS)

// Stage x_d[tstage] (64x512 f16) into the xt half of A1 with the CDNA5 async
// global->LDS DMA (GVS mode: saddr uniform base + per-lane 32-bit offset).
// Tracked by ASYNCcnt; retire with s_wait_asynccnt 0 before the next barrier.
__device__ __forceinline__ void stage_xt_async(const _Float16* __restrict__ xcur,
                                               int tstage, int tid) {
  const int row = tid >> 3, col0 = (tid & 7) * 64;     // 512 thr x 64 halves
  const _Float16* sbase = xcur + (size_t)tstage * Bn * Hn;   // uniform (SGPR)
  const unsigned goffb = (unsigned)((row * Hn + col0) * 2);  // per-lane bytes
  const unsigned ldsb  = (unsigned)((row * K1 + col0) * 2);  // LDS byte addr (A1 @ 0)
#pragma unroll
  for (int i = 0; i < 8; ++i) {
    asm volatile("global_load_async_to_lds_b128 %0, %1, %2"
                 :: "v"(ldsb + (unsigned)(i * 16)),
                    "v"(goffb + (unsigned)(i * 16)),
                    "s"(sbase)
                 : "memory");
  }
}

__global__ void seq_depth_kernel(const _Float16* __restrict__ xcur,
                                 _Float16* __restrict__ xnext,
                                 const _Float16* __restrict__ WU1t,
                                 const _Float16* __restrict__ WU2t,
                                 const float* __restrict__ bvec,   // (3H)
                                 const float* __restrict__ b_a1,   // (A)
                                 const float* __restrict__ Wa2,    // (A,2)
                                 const float* __restrict__ b_a2,   // (2)
                                 const float* __restrict__ mask,   // (B,T)
                                 const float* __restrict__ ap_prev, // (T,B) or unused
                                 const float* __restrict__ dm_prev, // (T,B) or unused
                                 float* __restrict__ act_ws,        // (T,B)
                                 float* __restrict__ dm_ws,         // (T,B)
                                 float* __restrict__ dout,
                                 int depth, int llm, int write_next) {
  extern __shared__ char smem[];
  _Float16* A1   = (_Float16*)smem;                 // 64 x 1024 (cols 0..511 xt, 512..1023 h)
  _Float16* RH   = A1 + Bn * K1;                    // 64 x 512 : r * h_tm1
  _Float16* ZB   = RH + Bn * Hn;                    // 64 x 512 : z
  float*    POLH = (float*)(ZB + Bn * Hn);          // 64 x 128
  float*    CB   = POLH + Bn * An;                  // both
  float*    CH   = CB + Bn;                         // h_only
  float*    CX   = CH + Bn;                         // x_only
  float*    CDMP = CX + Bn;                         // dmp
  float*    APRV = CDMP + Bn;                       // act carry
  float*    DMC  = APRV + Bn;                       // dm carry

  const int tid  = threadIdx.x;
  const int lane = tid & 31;
  const int wave = tid >> 5;          // 16 waves
  const int wm   = wave & 3;          // M-tile (batch rows)
  const int wg   = wave >> 2;         // N-tile group
  const int m0   = wm * 16;
  const float llmf = llm ? 1.f : 0.f;

  // init h = 0, carries = 0; kick off async stage of xt for t=0
  for (int i = tid; i < Bn * Hn; i += blockDim.x)
    A1[(size_t)(i >> 9) * K1 + Hn + (i & 511)] = (_Float16)0.f;
  if (tid < Bn) { APRV[tid] = 0.f; DMC[tid] = 0.f; }
  stage_xt_async(xcur, 0, tid);

  for (int t = 0; t < Tn; ++t) {
    // ---- retire async xt DMA; h(t-1) commits also ordered by this barrier ----
    asm volatile("s_wait_asynccnt 0x0" ::: "memory");
    __syncthreads();

    // ---- P1: GEMM1  [xt|h](64x1024) @ WU1t^T -> z | r (->r*h) | pol_h ----
    for (int it = 0; it < 18; ++it) {
      const int n0 = (wg * 18 + it) * 16;
      v8f acc = {};
#pragma unroll 4
      for (int kt = 0; kt < K1 / 32; ++kt) {
        v16h a = load_frag_a(A1, K1, m0, kt * 32, lane);
        v16h b = load_frag_b(WU1t, K1, n0, kt * 32, lane);
        WMMA_F16(acc, a, b);
      }
      const int col = n0 + (lane & 15);
      const int rbase = m0 + ((lane >> 4) << 3);
      if (n0 < Hn) {                       // z
        float bias = bvec[col];
#pragma unroll
        for (int j = 0; j < 8; ++j) {
          int row = rbase + j;
          ZB[(size_t)row * Hn + col] = (_Float16)hsig_f(acc[j] + bias);
        }
      } else if (n0 < 2 * Hn) {            // r -> r * h_tm1
        int c = col - Hn;
        float bias = bvec[col];
#pragma unroll
        for (int j = 0; j < 8; ++j) {
          int row = rbase + j;
          float r = hsig_f(acc[j] + bias);
          float h = (float)A1[(size_t)row * K1 + Hn + c];
          RH[(size_t)row * Hn + c] = (_Float16)(r * h);
        }
      } else {                             // pol_h (relu)
        int c = col - 2 * Hn;
        float bias = b_a1[c];
#pragma unroll
        for (int j = 0; j < 8; ++j) {
          int row = rbase + j;
          POLH[(size_t)row * An + c] = fmaxf(acc[j] + bias, 0.f);
        }
      }
    }
    __syncthreads();

    // ---- P2: policy head + per-batch act/gating logic ----
    if (tid < Bn) {
      const int b = tid;
      float p0 = b_a2[0], p1 = b_a2[1];
      for (int k = 0; k < An; ++k) {
        float ph = POLH[(size_t)b * An + k];
        p0 += ph * Wa2[2 * k];
        p1 += ph * Wa2[2 * k + 1];
      }
      p0 = expf(p0); p1 = expf(p1);

      float ap, dmp, dmp1;
      if (depth == 0) {
        ap = 0.f;
        dmp = mask[(size_t)b * Tn + t];
        dmp1 = t ? mask[(size_t)b * Tn + t - 1] : 1.f;
      } else {
        ap = (t < Tn - 1) ? ap_prev[(size_t)(t + 1) * Bn + b] : 0.f;
        dmp = dm_prev[(size_t)t * Bn + b];
        dmp1 = t ? dm_prev[(size_t)(t - 1) * Bn + b] : 1.f;
      }
      float e1 = t ? mask[(size_t)b * Tn + t - 1] * (1.f - mask[(size_t)b * Tn + t]) : 0.f;

      float act = (p0 >= p1) ? 1.f : 0.f;
      if (ap > 0.f) act = 1.f;
      if (llm)      act = 1.f;
      if (e1 > 0.f) act = 0.f;
      float ac = dmp1 * (1.f - llmf) * (1.f - e1) * (1.f - ap);

      float dmc = DMC[b], a_tm1 = APRV[b];
      float both  = (1.f - ap) * dmp * act * dmc;
      float honly = dmc * act * (ap + (1.f - ap) * (1.f - dmp));
      float xonly = dmp * (1.f - ap) * (1.f - act + act * (1.f - dmc));
      float dmn   = dmp * (both + xonly + honly);
      float act_o = (dmp1 > 0.f) ? act : a_tm1;

      CB[b] = both; CH[b] = honly; CX[b] = xonly; CDMP[b] = dmp;
      DMC[b] = dmn; APRV[b] = act_o;

      act_ws[(size_t)t * Bn + b] = act_o;
      dm_ws[(size_t)t * Bn + b]  = dmn;

      size_t base = (size_t)(b * NDEP + depth) * Tn + t;
      dout[OUT_AC + base] = ac;
      dout[OUT_POL + base * 2]     = p0;
      dout[OUT_POL + base * 2 + 1] = p1;
      if (t > 0) dout[OUT_ACTION + base - 1] = act_o;  // sa[t-1] = act[t]
    }
    __syncthreads();

    // ---- P3: GEMM2 [xt|r*h](64x1024) @ WU2t^T -> h_cand; fuse GRU combine ----
    float hs[8][8];
#pragma unroll
    for (int it = 0; it < 8; ++it) {
      const int n0 = (wg * 8 + it) * 16;
      v8f acc = {};
#pragma unroll 4
      for (int kt = 0; kt < K1 / 32; ++kt) {
        v16h a = (kt < 16) ? load_frag_a(A1, K1, m0, kt * 32, lane)
                           : load_frag_a(RH, Hn, m0, (kt - 16) * 32, lane);
        v16h b = load_frag_b(WU2t, K1, n0, kt * 32, lane);
        WMMA_F16(acc, a, b);
      }
      const int col = n0 + (lane & 15);
      const int rbase = m0 + ((lane >> 4) << 3);
      const float bias = bvec[2 * Hn + col];
#pragma unroll
      for (int j = 0; j < 8; ++j) {
        int row = rbase + j;
        float hcv = tanhf(acc[j] + bias);
        float z   = (float)ZB[(size_t)row * Hn + col];
        float h   = (float)A1[(size_t)row * K1 + Hn + col];
        float xv  = (float)A1[(size_t)row * K1 + col];
        float hc  = z * h + (1.f - z) * hcv;
        float hn  = (CB[row] > 0.f) ? hc : 0.f;
        if (CH[row] > 0.f) hn = h;
        if (CX[row] > 0.f) hn = xv;
        hn = (CDMP[row] > 0.f) ? hn : h;
        hs[it][j] = hn;
      }
    }
    __syncthreads();  // all xt / h_tm1 reads complete

    // ---- P4: launch async stage of next xt (overlaps stores), commit h ----
    if (t + 1 < Tn) stage_xt_async(xcur, t + 1, tid);

#pragma unroll
    for (int it = 0; it < 8; ++it) {
      const int n0 = (wg * 8 + it) * 16;
      const int col = n0 + (lane & 15);
      const int rbase = m0 + ((lane >> 4) << 3);
#pragma unroll
      for (int j = 0; j < 8; ++j) {
        int row = rbase + j;
        float hn = hs[it][j];
        A1[(size_t)row * K1 + Hn + col] = (_Float16)hn;
        size_t pidx = ((size_t)(row * NDEP + depth) * Tn + t) * Hn + col;
        dout[OUT_PIH + pidx] = (t == 0) ? 0.f : hn;  // pih zeroes t=0
        if (write_next) {
          xnext[((size_t)t * Bn + row) * Hn + col] = (_Float16)hn;
          dout[OUT_PIX + ((size_t)(row * NDEP + depth + 1) * Tn + t) * Hn + col] = hn;
        }
        if (depth == NDEP - 1 && t == Tn - 1)
          dout[OUT_OUTPUT + (size_t)row * Hn + col] = hn;  // hS[-1,-1]
      }
    }
    // loop-top wait+barrier orders the async DMA and h commits vs P1 reads
  }

  if (tid < Bn)  // sa[T-1] = 0
    dout[OUT_ACTION + (size_t)(tid * NDEP + depth) * Tn + (Tn - 1)] = 0.f;
}

// ---------------------------------------------------------------------------
extern "C" void kernel_launch(void* const* d_in, const int* in_sizes, int n_in,
                              void* d_out, int out_size, void* d_ws, size_t ws_size,
                              hipStream_t stream) {
  (void)in_sizes; (void)n_in; (void)out_size; (void)ws_size;

  const float* x     = (const float*)d_in[0];
  const float* mask  = (const float*)d_in[1];
  const float* Wemb  = (const float*)d_in[3];
  const float* b_emb = (const float*)d_in[4];
  const float* W     = (const float*)d_in[5];
  const float* U     = (const float*)d_in[6];
  const float* bvec  = (const float*)d_in[7];
  const float* Wa1   = (const float*)d_in[8];
  const float* Ua1   = (const float*)d_in[9];
  const float* b_a1  = (const float*)d_in[10];
  const float* Wa2   = (const float*)d_in[11];
  const float* b_a2  = (const float*)d_in[12];
  float* dout = (float*)d_out;

  char* ws = (char*)d_ws;
  _Float16* WU1t = (_Float16*)(ws + WS_WU1T);
  _Float16* WU2t = (_Float16*)(ws + WS_WU2T);
  _Float16* Wet  = (_Float16*)(ws + WS_WET);
  _Float16* xh   = (_Float16*)(ws + WS_XH);
  _Float16* xd[2] = { (_Float16*)(ws + WS_XD0), (_Float16*)(ws + WS_XD1) };
  float* actb[2]  = { (float*)(ws + WS_ACT0), (float*)(ws + WS_ACT1) };
  float* dmb[2]   = { (float*)(ws + WS_DM0),  (float*)(ws + WS_DM1)  };

  cvt_x_kernel<<<2048, 512, 0, stream>>>(x, xh, Bn * Tn * DIN);
  build_w_kernel<<<2048, 512, 0, stream>>>(W, U, Wa1, Ua1, Wemb, WU1t, WU2t, Wet);
  // xe: (B*T, DIN) @ (DIN, H); 1024 x 32 tiles, one wave per tile
  embed_kernel<<<4096, 256, 0, stream>>>(xh, Wet, b_emb, xd[0], dout);

  for (int d = 0; d < NDEP; ++d) {
    seq_depth_kernel<<<1, 512, SEQ_LDS, stream>>>(
        xd[d & 1], xd[(d + 1) & 1], WU1t, WU2t,
        bvec, b_a1, Wa2, b_a2, mask,
        d ? actb[(d - 1) & 1] : mask,   // unused when depth==0
        d ? dmb[(d - 1) & 1]  : mask,   // unused when depth==0
        actb[d & 1], dmb[d & 1], dout,
        d, (d == NDEP - 1) ? 1 : 0, (d < NDEP - 1) ? 1 : 0);
  }
}
